// NonLocalMean_15633680957733
// MI455X (gfx1250) — matrix operand order repaired and checked by reading, not compiled, not run on hardware
//
#include <hip/hip_runtime.h>
#include <hip/hip_bf16.h>

typedef __attribute__((ext_vector_type(16))) _Float16 v16h;
typedef __attribute__((ext_vector_type(8)))  _Float16 v8h;
typedef __attribute__((ext_vector_type(4)))  _Float16 v4h;
typedef __attribute__((ext_vector_type(8)))  float    v8f;

#define KW     7      // window size
#define PADR   3      // (KW-1)/2
#define TW     16     // pixel tile width (WMMA M)
#define WPAD   32     // staged halo width (cols c <-> image col w0-4+c)
#define GHEADS 4
#define MID    16     // channels per head
#define CCH    64
#define HH     128
#define WWID   128

// Band mapping: center pixel m sits at staged col m+4; window offset dj(0..6)
// is staged col m+dj+1 (range 1..22). Staged cols are 16B-aligned float4
// chunks of the image row, so chunks never straddle the image boundary.

// One block = 128 threads = 4 wave32 = 4 heads; handles one (b, h, 16-wide tile).
__global__ __launch_bounds__(128)
void nlm7x7_wmma_kernel(const float* __restrict__ x, float* __restrict__ out) {
    // channel-fastest copy: [row][col][ch]  (correlation A/B operands)
    __shared__ __align__(32) _Float16 sTileC[KW * WPAD * CCH];     // 28672 B
    // column-fastest copy:  [ch][row][col]  (aggregation B operand)
    __shared__ __align__(32) _Float16 sTileW[CCH * KW * WPAD];     // 28672 B
    __shared__ float    sNorm[GHEADS * KW * WPAD];                 //  3584 B
    __shared__ _Float16 sS[GHEADS * KW * TW * WPAD];               // 28672 B
    __shared__ _Float16 sW[GHEADS * TW * 56];                      //  7168 B
    __shared__ float    sY[GHEADS * TW * TW];                      //  4096 B
    __shared__ __align__(32) _Float16 sZeroH[32];                  //    64 B
    __shared__ _Float16 sDumpH[64];                                //   128 B

    const int tid = threadIdx.x;
    const int b   = blockIdx.z;
    const int h   = blockIdx.y;
    const int w0  = blockIdx.x * TW;

    if (tid < 32) sZeroH[tid] = (_Float16)0.0f;

    // ---- Phase 1: stage halo (float4 chunks, branch-free, dual layout) ----
    // chunks: ch(64) x row(7) x q(8); chunk q covers staged cols 4q..4q+3.
    for (int i = tid; i < CCH * KW * 8; i += 128) {
        int ch  = i / (KW * 8);
        int rem = i % (KW * 8);
        int row = rem / 8;
        int q   = rem % 8;
        int hh  = h + row - PADR;
        int ww0 = w0 - 4 + 4 * q;
        bool inb = (hh >= 0) & (hh < HH) & (ww0 >= 0) & (ww0 <= WWID - 4);
        int hc = min(max(hh, 0), HH - 1);
        int wc = min(max(ww0, 0), WWID - 4);
        const float4 v =
            *reinterpret_cast<const float4*>(x + (((b * CCH + ch) * HH + hc) * WWID + wc));
        v4h hv;
        hv[0] = (_Float16)(inb ? v.x : 0.0f);
        hv[1] = (_Float16)(inb ? v.y : 0.0f);
        hv[2] = (_Float16)(inb ? v.z : 0.0f);
        hv[3] = (_Float16)(inb ? v.w : 0.0f);
        // column-fastest store: 4 contiguous halves (one b64 store)
        *reinterpret_cast<v4h*>(&sTileW[(ch * KW + row) * WPAD + 4 * q]) = hv;
        // channel-fastest stores: strided scalar halves
        #pragma unroll
        for (int t = 0; t < 4; ++t)
            sTileC[((row * WPAD + 4 * q + t) * CCH) + ch] = hv[t];
    }
    __syncthreads();

    // ---- Phase 2: per-head squared norms (vectorized channel reads) -------
    for (int i = tid; i < GHEADS * KW * WPAD; i += 128) {
        int g   = i / (KW * WPAD);
        int rem = i % (KW * WPAD);             // row*WPAD + col
        const v16h t = *reinterpret_cast<const v16h*>(&sTileC[rem * CCH + g * MID]);
        float s = 0.0f;
        #pragma unroll
        for (int c = 0; c < MID; ++c) {
            float v = (float)t[c];
            s += v * v;
        }
        sNorm[i] = s;
    }
    __syncthreads();

    const int lane = tid & 31;
    const int g    = tid >> 5;     // wave == head
    const int half = lane >> 4;    // upper/lower 16-lane group
    const int mn   = lane & 15;    // pixel (as M) or column (as N) index

    // ---- Phase 3: correlation S = <center, neighbor> (vector operands) ----
    // A operand (contiguous k order used here): element j -> K = j + 16*half.
    v16h a;
    {
        const v16h alo = *reinterpret_cast<const v16h*>(
            &sTileC[((PADR * WPAD + mn + 4) * CCH) + g * MID]);
        #pragma unroll
        for (int j = 0; j < 16; ++j)
            a[j] = (half == 0) ? alo[j] : (_Float16)0.0f;
    }

    const bool loN = (half == 0);  // lanes holding K=0..15 of B matrices
    #pragma unroll
    for (int dh = 0; dh < KW; ++dh) {
        // B (same contiguous k order as A): element j -> K = j + 16*half (channel), N = lane%16 (col).
        const v16h* p0 = loN
            ? reinterpret_cast<const v16h*>(&sTileC[((dh * WPAD + mn) * CCH) + g * MID])
            : reinterpret_cast<const v16h*>(&sZeroH[0]);
        const v16h* p1 = loN
            ? reinterpret_cast<const v16h*>(&sTileC[((dh * WPAD + 16 + mn) * CCH) + g * MID])
            : reinterpret_cast<const v16h*>(&sZeroH[0]);
        v16h b0 = *p0;
        v16h b1 = *p1;
        v8f z = {};
        v8f d0 = __builtin_amdgcn_wmma_f32_16x16x32_f16(false, a, false, b0,
                                                        (short)0, z, false, false);
        v8f d1 = __builtin_amdgcn_wmma_f32_16x16x32_f16(false, a, false, b1,
                                                        (short)0, z, false, false);
        // D layout: element r -> (M = r + 8*half, N = lane%16). Unconditional spill.
        #pragma unroll
        for (int r = 0; r < 8; ++r) {
            int m2 = r + 8 * half;
            sS[((g * KW + dh) * TW + m2) * WPAD + mn]      = (_Float16)d0[r];
            sS[((g * KW + dh) * TW + m2) * WPAD + 16 + mn] = (_Float16)d1[r];
        }
    }
    __syncthreads();

    // ---- Phase 4: distances + 49-way softmax (branch-free) ----------------
    {
        const int m = mn;                       // both lane halves: pixel m
        const float normC = sNorm[(g * KW + PADR) * WPAD + m + 4];
        float dv[28];
        #pragma unroll
        for (int it = 0; it < 4; ++it) {
            int dh = (half ? 4 : 0) + it;       // half 0: 0..3, half 1: 4..7
            bool valid = dh < KW;
            int dhc = valid ? dh : (KW - 1);
            #pragma unroll
            for (int dj = 0; dj < 7; ++dj) {
                int col = m + dj + 1;           // staged col of neighbor
                float s = (float)sS[((g * KW + dhc) * TW + m) * WPAD + col];
                float d = sNorm[(g * KW + dhc) * WPAD + col] + normC - 2.0f * s;
                dv[it * 7 + dj] = valid ? d : 3.4e38f;   // => weight 0
            }
        }
        float dmin = dv[0];
        #pragma unroll
        for (int i = 1; i < 28; ++i) dmin = fminf(dmin, dv[i]);
        dmin = fminf(dmin, __shfl_xor(dmin, 16));        // wave32 cross-half

        float sum = 0.0f;
        #pragma unroll
        for (int i = 0; i < 28; ++i) {
            float e = __expf(dmin - dv[i]);              // invalid -> exp(-inf)=0
            dv[i] = e;
            sum += e;
        }
        sum += __shfl_xor(sum, 16);
        float inv = 1.0f / sum;

        #pragma unroll
        for (int it = 0; it < 4; ++it) {
            int dh = (half ? 4 : 0) + it;
            bool valid = dh < KW;
            int dhc = valid ? dh : (KW - 1);
            #pragma unroll
            for (int dj = 0; dj < 7; ++dj) {
                _Float16* q = valid ? &sW[(g * TW + m) * 56 + dhc * 8 + dj]
                                    : &sDumpH[lane];
                *q = (_Float16)(dv[it * 7 + dj] * inv);
            }
        }
    }
    __syncthreads();

    // ---- Phase 5: aggregation Y = W_banded x Tile (WMMA) ------------------
    v8f acc = {};
    #pragma unroll
    for (int dh = 0; dh < KW; ++dh) {
        v16h a2;
        #pragma unroll
        for (int j = 0; j < 16; ++j) {
            // A: M = pixel (mn), K = staged column; banded weights.
            int kc = j + 16 * half;   // staged col 0..31
            unsigned t = (unsigned)(kc - mn - 1);         // dj offset
            const _Float16* pa = (t < 7u)
                ? &sW[(g * TW + mn) * 56 + dh * 8 + (int)t] : &sZeroH[0];
            a2[j] = *pa;
        }
        // B: K = staged col (16*half..16*half+15 contiguous), N = channel (mn).
        // Weight band is zero for cols >= 23, so dummy cols contribute nothing.
        const v16h b2 = *reinterpret_cast<const v16h*>(
            &sTileW[((g * MID + mn) * KW + dh) * WPAD + 16 * half]);
        acc = __builtin_amdgcn_wmma_f32_16x16x32_f16(false, a2, false, b2,
                                                     (short)0, acc, false, false);
    }
    #pragma unroll
    for (int r = 0; r < 8; ++r)
        sY[g * 256 + (r + 8 * half) * TW + mn] = acc[r];   // [pixel][channel]
    __syncthreads();

    // ---- Phase 6: residual add + coalesced store --------------------------
    for (int i = lane; i < 256; i += 32) {
        int ch = i >> 4;
        int m  = i & 15;
        int gi = ((b * CCH + g * MID + ch) * HH + h) * WWID + w0 + m;
        out[gi] = x[gi] + sY[g * 256 + m * TW + ch];
    }
}

extern "C" void kernel_launch(void* const* d_in, const int* in_sizes, int n_in,
                              void* d_out, int out_size, void* d_ws, size_t ws_size,
                              hipStream_t stream) {
    (void)in_sizes; (void)n_in; (void)d_ws; (void)ws_size; (void)out_size;
    const float* x = (const float*)d_in[0];
    float* out = (float*)d_out;
    dim3 grid(WWID / TW, HH, 4);   // (8, 128, 4) tiles
    dim3 block(128);               // 4 wave32 = 4 heads
    nlm7x7_wmma_kernel<<<grid, block, 0, stream>>>(x, out);
}
